// EncoderLSTM_15719580303481
// MI455X (gfx1250) — compile-verified, matrix operand order checked by reading, not run on hardware
//
#include <hip/hip_runtime.h>
#include <hip/hip_bf16.h>

#define B_   16384
#define H_   1024
#define T_   60
#define IN_  4
#define G4H  (4 * H_)          // 4096
#define NL   (G4H * H_)        // W_hh elements per layer

typedef __attribute__((ext_vector_type(16))) __bf16        v16bf;
typedef __attribute__((ext_vector_type(8)))  __bf16        v8bf;
typedef __attribute__((ext_vector_type(8)))  float         v8f;
typedef __attribute__((ext_vector_type(4)))  unsigned int  u32x4;

__device__ __forceinline__ unsigned short f2bf(float f) {
    unsigned int u = __float_as_uint(f);
    u = (u + 0x7FFFu + ((u >> 16) & 1u)) >> 16;   // round-to-nearest-even
    return (unsigned short)u;
}
__device__ __forceinline__ float sigf(float x)     { return 1.f / (1.f + __expf(-x)); }
__device__ __forceinline__ float tanhfast(float x) { return 1.f - 2.f / (__expf(2.f * x) + 1.f); }

__device__ __forceinline__ v16bf frag_from(const unsigned short* p) {
    // 16B-aligned; low half = K..K+7, high half (+32B) = K+16..K+23
    const v8bf* vp = reinterpret_cast<const v8bf*>(p);
    v8bf lo = vp[0], hi = vp[2];
    v16bf r;
    #pragma unroll
    for (int i = 0; i < 8; ++i) { r[i] = lo[i]; r[i + 8] = hi[i]; }
    return r;
}

// ---------------------------------------------------------------- utilities
__global__ void zero_u32(unsigned int* p, long n) {
    long i = (long)blockIdx.x * blockDim.x + threadIdx.x;
    long s = (long)gridDim.x * blockDim.x;
    for (; i < n; i += s) p[i] = 0u;
}

__global__ void convert_bf16(const float* __restrict__ src, unsigned short* __restrict__ dst, int n) {
    int i = blockIdx.x * blockDim.x + threadIdx.x;
    int s = gridDim.x * blockDim.x;
    for (; i < n; i += s) dst[i] = f2bf(src[i]);
}

// ---------------------------------------------------------------- LSTM cell
// Grid: (B/128, H/32), block: 256 threads (8 waves).
// Each WG: rows [bx*128, +128), hidden cols [by*32, +32), all 4 gates.
// Wave w owns rows [bx*128 + w*16, +16) and computes 8 n-tiles:
//   q = gate*2 + ht  -> gate-col block  gate*H + by*32 + ht*16
__global__ __launch_bounds__(256) void lstm_cell_kernel(
    const float* __restrict__ x,            // [B][IN][T]
    int t,
    const unsigned short* __restrict__ Whh, // bf16 [4H][H]
    const float* __restrict__ Wih,          // [4H][IN]
    const float* __restrict__ bih,
    const float* __restrict__ bhh,
    const unsigned short* __restrict__ hin, // bf16 [B][H]
    unsigned short* __restrict__ hout,      // bf16 [B][H]
    float* __restrict__ Cst,                // f32 [B][H], updated in place
    float* __restrict__ outF)               // f32 [B][H] final h, or nullptr
{
    // double-buffered B tile: 2 x [8 q-tiles][16 rows][64 K] bf16 = 32 KB
    __shared__ __attribute__((aligned(16))) unsigned char ldsB[2 * 8 * 16 * 64 * 2];

    const int lane = threadIdx.x & 31;
    const int wave = threadIdx.x >> 5;
    const int mBase    = blockIdx.x * 128 + wave * 16;
    const int nHidBase = blockIdx.y * 32;

    // ---- this wave's staging sub-tile of W_hh (q == wave) -----------------
    const int sg  = wave >> 1;                    // gate
    const int sht = wave & 1;                     // hidden half-tile
    const int sj  = sg * H_ + nHidBase + sht * 16 + (lane >> 1);   // W row
    const int skoff = (lane & 1) * 32;            // K offset within chunk
    const int ldsStoreOff = wave * 2048 + lane * 64;

    v8f acc[8];
    #pragma unroll
    for (int q = 0; q < 8; ++q) acc[q] = (v8f)0.f;

    const int kh8  = (lane >> 4) * 8;             // K sub-offset for hi lanes
    const int arow = (mBase + (lane & 15)) * H_;  // A row base (u16 units)
    const int bRowOff = (lane & 15) * 128 + (lane >> 4) * 16; // B frag row bytes

    // stage chunk 0 into buffer 0
    {
        const u32x4* g = reinterpret_cast<const u32x4*>(Whh + sj * H_ + skoff);
        u32x4 s0 = g[0], s1 = g[1], s2 = g[2], s3 = g[3];
        u32x4* lp = reinterpret_cast<u32x4*>(&ldsB[ldsStoreOff]);
        lp[0] = s0; lp[1] = s1; lp[2] = s2; lp[3] = s3;
    }
    __syncthreads();

    int p = 0;
    for (int kc = 0; kc < H_; kc += 64) {
        const bool more = (kc + 64) < H_;
        u32x4 s0, s1, s2, s3;
        if (more) {
            const u32x4* g =
                reinterpret_cast<const u32x4*>(Whh + sj * H_ + (kc + 64) + skoff);
            s0 = g[0]; s1 = g[1]; s2 = g[2]; s3 = g[3];
            __builtin_prefetch(hin + arow + kc + 64 + kh8, 0, 1); // next A chunk
        }

        // both A fragments for this chunk up front (global, L2-resident)
        v16bf a0 = frag_from(hin + arow + kc + kh8);
        v16bf a1 = frag_from(hin + arow + kc + 32 + kh8);

        // 16 WMMAs, B-fragment loads pipelined one step ahead
        const unsigned char* bufBase = &ldsB[p * 16384 + bRowOff];
        v16bf bcur = frag_from(reinterpret_cast<const unsigned short*>(bufBase));
        #pragma unroll
        for (int k16 = 0; k16 < 16; ++k16) {
            v16bf bnxt;
            if (k16 < 15) {
                const int s = (k16 + 1) >> 3;
                const int q = (k16 + 1) & 7;
                bnxt = frag_from(reinterpret_cast<const unsigned short*>(
                    bufBase + q * 2048 + s * 64));
            }
            const int qa = k16 & 7;
            acc[qa] = __builtin_amdgcn_wmma_f32_16x16x32_bf16(
                false, (k16 < 8) ? a0 : a1, false, bcur, (short)0, acc[qa],
                false, false);
            bcur = bnxt;
        }

        if (more) {   // stage next chunk into the other buffer
            u32x4* lp = reinterpret_cast<u32x4*>(&ldsB[(p ^ 1) * 16384 + ldsStoreOff]);
            lp[0] = s0; lp[1] = s1; lp[2] = s2; lp[3] = s3;
        }
        __syncthreads();
        p ^= 1;
    }

    // ---- epilogue: gates -> cell update ----------------------------------
    // C/D layout: lane l, vgpr v -> M = v + 8*(l>>4), N = l&15
    const int nloc  = lane & 15;
    const int mrow0 = mBase + ((lane >> 4) << 3);

    #pragma unroll
    for (int ht = 0; ht < 2; ++ht) {
        const int n = nHidBase + ht * 16 + nloc;          // hidden column
        float bsum[4], wv[4][4];
        #pragma unroll
        for (int g = 0; g < 4; ++g) {
            const int j = g * H_ + n;
            bsum[g] = bih[j] + bhh[j];
            #pragma unroll
            for (int i = 0; i < IN_; ++i) wv[g][i] = Wih[j * IN_ + i];
        }
        #pragma unroll
        for (int v = 0; v < 8; ++v) {
            const int m = mrow0 + v;
            float xm[4];
            #pragma unroll
            for (int i = 0; i < IN_; ++i) xm[i] = x[m * (IN_ * T_) + i * T_ + t];
            float dots[4];
            #pragma unroll
            for (int g = 0; g < 4; ++g)
                dots[g] = wv[g][0] * xm[0] + wv[g][1] * xm[1] +
                          wv[g][2] * xm[2] + wv[g][3] * xm[3];

            const float gi = acc[0 * 2 + ht][v] + bsum[0] + dots[0];
            const float gf = acc[1 * 2 + ht][v] + bsum[1] + dots[1];
            const float gg = acc[2 * 2 + ht][v] + bsum[2] + dots[2];
            const float go = acc[3 * 2 + ht][v] + bsum[3] + dots[3];

            const int cidx = m * H_ + n;
            const float cold = Cst[cidx];
            const float c2 = sigf(gf) * cold + sigf(gi) * tanhfast(gg);
            const float h2 = sigf(go) * tanhfast(c2);
            Cst[cidx]  = c2;
            hout[cidx] = f2bf(h2);
            if (outF) outF[cidx] = h2;
        }
    }
}

// ---------------------------------------------------------------- launcher
extern "C" void kernel_launch(void* const* d_in, const int* in_sizes, int n_in,
                              void* d_out, int out_size, void* d_ws, size_t ws_size,
                              hipStream_t stream) {
    (void)in_sizes; (void)n_in; (void)out_size; (void)ws_size;

    const float* input = (const float*)d_in[0];
    // per-layer params: d_in[1+4L]=W_ih, [2+4L]=W_hh, [3+4L]=b_ih, [4+4L]=b_hh

    unsigned char* ws = (unsigned char*)d_ws;
    unsigned short* Wb = (unsigned short*)ws;                         // 3*NL bf16 (24 MB)
    size_t off = (size_t)3 * NL * 2;
    unsigned short* hA = (unsigned short*)(ws + off);                 // 32 MB
    off += (size_t)B_ * H_ * 2;
    float* Cst = (float*)(ws + off);                                  // 64 MB
    off += (size_t)B_ * H_ * 4;
    unsigned short* hB = (unsigned short*)(ws + off);                 // 32 MB

    // zero hA and Cst (contiguous 96 MB region)
    {
        long words = ((long)B_ * H_ * 2 + (long)B_ * H_ * 4) / 4;
        zero_u32<<<4096, 256, 0, stream>>>((unsigned int*)hA, words);
    }
    // convert W_hh of each layer to bf16
    for (int L = 0; L < 3; ++L) {
        convert_bf16<<<2048, 256, 0, stream>>>((const float*)d_in[2 + 4 * L],
                                               Wb + (size_t)L * NL, NL);
    }

    dim3 grid(B_ / 128, H_ / 32);
    for (int t = 0; t < T_; ++t) {
        for (int L = 0; L < 3; ++L) {
            const int cell = t * 3 + L;
            const unsigned short* hin = (cell & 1) ? hB : hA;
            unsigned short* houtp     = (cell & 1) ? hA : hB;
            const bool last = (cell == 3 * T_ - 1);
            lstm_cell_kernel<<<grid, 256, 0, stream>>>(
                input, t,
                Wb + (size_t)L * NL,
                (const float*)d_in[1 + 4 * L],
                (const float*)d_in[3 + 4 * L],
                (const float*)d_in[4 + 4 * L],
                hin, houtp, Cst,
                last ? (float*)d_out : nullptr);
        }
    }
}
